// MambaPredictor_14963666059501
// MI455X (gfx1250) — compile-verified, matrix operand order checked by reading, not run on hardware
//
#include <hip/hip_runtime.h>
#include <cstdint>
#include <type_traits>

#define B_SZ     8
#define L_SZ     1024
#define IN_DIM_  256
#define D_MODEL_ 1024
#define D_STATE_ 16
#define D_CONV_  4
#define D_INNER_ 2048
#define DT_RANK_ 64
#define N_CLS_   1000
#define XP_DIM_  (DT_RANK_ + 2 * D_STATE_)   // 96
#define M_ROWS   (B_SZ * L_SZ)               // 8192
#define NCHUNK_  16
#define CH_LEN_  (L_SZ / NCHUNK_)            // 64

typedef __bf16 bf16_t;
typedef __attribute__((ext_vector_type(16))) __bf16 v16bf;
typedef __attribute__((ext_vector_type(8)))  float  v8f;

// ---------------------------------------------------------------- helpers
__device__ __forceinline__ float silu_f(float x) {
    return x / (1.0f + __expf(-x));
}
__device__ __forceinline__ float softplus_f(float x) {
    return (x > 20.0f) ? x : log1pf(__expf(x));
}

template <typename OutT>
__device__ __forceinline__ void st_elem(OutT* p, float v) {
    if constexpr (std::is_same<OutT, float>::value) {
        *p = v;
    } else {
        *p = (bf16_t)v;
    }
}

// ---------------------------------------------------------------- f32 -> bf16
__global__ __launch_bounds__(256) void cvt_f32_bf16(const float* __restrict__ in,
                                                    bf16_t* __restrict__ out,
                                                    size_t n) {
    size_t i = (size_t)blockIdx.x * blockDim.x + threadIdx.x;
    if (i < n) out[i] = (bf16_t)in[i];
}

// ---------------------------------------------------------------- WMMA GEMM
// C[M,N] = act( A[M,K] * W[N,K]^T + bias )   (A, W bf16 row-major; f32 accum)
// Wave tile: 32 x (TN*16); 2*TN wmma per K-step on (2+TN) fragments.
template <typename OutT, bool HAS_BIAS, int ACT, int TN>
__global__ __launch_bounds__(256) void wmma_gemm_nt(const bf16_t* __restrict__ A,
                                                    const bf16_t* __restrict__ W,
                                                    const float* __restrict__ bias,
                                                    OutT* __restrict__ C,
                                                    int M, int N, int K) {
    const int lane   = threadIdx.x & 31;
    const int wave   = threadIdx.x >> 5;
    const int tilesN = N / (TN * 16);
    const int flat   = blockIdx.x * (blockDim.x >> 5) + wave;
    const int tm     = flat / tilesN;
    const int tn     = flat - tm * tilesN;
    if (tm * 32 >= M) return;                 // uniform whole-wave exit

    const int row0 = tm * 32;
    const int col0 = tn * (TN * 16);

    // Fragment addressing (identical pattern for A and B fragments):
    // lane&15 selects the matrix row; lane>>4 selects K half [0..15] vs [16..31].
    const int fr    = lane & 15;
    const int kHalf = (lane >> 4) * 16;

    const bf16_t* Ap[2];
    Ap[0] = A + (size_t)(row0 + fr) * K + kHalf;
    Ap[1] = Ap[0] + (size_t)16 * K;
    const bf16_t* Bp[TN];
#pragma unroll
    for (int j = 0; j < TN; ++j)
        Bp[j] = W + (size_t)(col0 + j * 16 + fr) * K + kHalf;

    v8f acc[2][TN];
#pragma unroll
    for (int i = 0; i < 2; ++i)
#pragma unroll
        for (int j = 0; j < TN; ++j) acc[i][j] = (v8f){};

    for (int k = 0; k < K; k += 32) {
        if (k + 64 < K) {                      // global_prefetch_b8 next tiles
            __builtin_prefetch(Ap[0] + k + 64, 0, 1);
            __builtin_prefetch(Bp[0] + k + 64, 0, 1);
        }
        v16bf af[2];
        af[0] = *(const v16bf*)(Ap[0] + k);
        af[1] = *(const v16bf*)(Ap[1] + k);
        v16bf bw[TN];
#pragma unroll
        for (int j = 0; j < TN; ++j) bw[j] = *(const v16bf*)(Bp[j] + k);
#pragma unroll
        for (int i = 0; i < 2; ++i)
#pragma unroll
            for (int j = 0; j < TN; ++j)
                acc[i][j] = __builtin_amdgcn_wmma_f32_16x16x32_bf16(
                    false, af[i], false, bw[j], (short)0, acc[i][j], false, false);
    }

    // C fragment: VGPR e -> M = e + 8*(lane>=16), N = lane&15
    const int cn = lane & 15;
    const int ma = (lane >> 4) * 8;
    float bs[TN];
#pragma unroll
    for (int j = 0; j < TN; ++j)
        bs[j] = HAS_BIAS ? bias[col0 + j * 16 + cn] : 0.0f;

#pragma unroll
    for (int i = 0; i < 2; ++i) {
#pragma unroll
        for (int e = 0; e < 8; ++e) {
            const int r = row0 + i * 16 + ma + e;
#pragma unroll
            for (int j = 0; j < TN; ++j) {
                float v = acc[i][j][e] + bs[j];
                if constexpr (ACT == 1) v = softplus_f(v);
                st_elem(C + (size_t)r * N + col0 + j * 16 + cn, v);
            }
        }
    }
}

// ---------------------------------------------------------------- z at t=L-1 only
__global__ __launch_bounds__(256) void zlast_kernel(const bf16_t* __restrict__ emb16,
                                                    const float* __restrict__ W_in,
                                                    float* __restrict__ zlast) {
    int idx = blockIdx.x * blockDim.x + threadIdx.x;     // B*D_INNER
    if (idx >= B_SZ * D_INNER_) return;
    const int d = idx & (D_INNER_ - 1);
    const int b = idx >> 11;
    const bf16_t* e = emb16 + (size_t)(b * L_SZ + (L_SZ - 1)) * D_MODEL_;
    const float*  w = W_in + (size_t)(D_INNER_ + d) * D_MODEL_;  // second half rows
    float acc = 0.0f;
#pragma unroll 4
    for (int m = 0; m < D_MODEL_; ++m) acc += (float)e[m] * w[m];
    zlast[idx] = acc;
}

// ---------------------------------------------------------------- depthwise causal conv + SiLU
__global__ __launch_bounds__(256) void conv_silu_kernel(const float* __restrict__ xs,
                                                        const float* __restrict__ Wc,
                                                        const float* __restrict__ bc,
                                                        bf16_t* __restrict__ u16) {
    size_t idx = (size_t)blockIdx.x * blockDim.x + threadIdx.x;  // over M_ROWS*D_INNER
    if (idx >= (size_t)M_ROWS * D_INNER_) return;
    const int d  = (int)(idx & (D_INNER_ - 1));
    const int bl = (int)(idx >> 11);
    const int l  = bl & (L_SZ - 1);
    const int b  = bl >> 10;
    float acc = bc[d];
#pragma unroll
    for (int k = 0; k < D_CONV_; ++k) {
        const int t = l - (D_CONV_ - 1) + k;
        if (t >= 0)
            acc += xs[(size_t)(b * L_SZ + t) * D_INNER_ + d] * Wc[d * D_CONV_ + k];
    }
    u16[idx] = (bf16_t)silu_f(acc);
}

// ---------------------------------------------------------------- slice dt cols -> bf16
__global__ __launch_bounds__(256) void extract_dt_kernel(const float* __restrict__ xdbl,
                                                         bf16_t* __restrict__ dt16) {
    size_t i = (size_t)blockIdx.x * blockDim.x + threadIdx.x;    // M_ROWS*DT_RANK
    if (i >= (size_t)M_ROWS * DT_RANK_) return;
    const size_t r = i >> 6;
    const int    c = (int)(i & (DT_RANK_ - 1));
    dt16[i] = (bf16_t)xdbl[r * XP_DIM_ + c];
}

// ---------------------------------------------------------------- selective scan, pass 1
// Diagonal linear recurrence h_t = dA_t * h + dBu_t  =>  per-chunk transition is
// the elementwise product P = prod(dA_t). Each block handles one (b, chunk,
// 256-channel slice): computes (P, h_local) over CH_LEN_ steps with state in
// registers. B_t (16 floats, shared by all channels) is staged into LDS with
// global_load_async_to_lds_b128 (ASYNCcnt path).
__global__ __launch_bounds__(256) void scan_pass1(const float* __restrict__ delta,
                                                  const float* __restrict__ xdbl,
                                                  const bf16_t* __restrict__ u16,
                                                  const float* __restrict__ A_log,
                                                  float* __restrict__ Pbuf,
                                                  float* __restrict__ Hbuf) {
    __shared__ float sB[D_STATE_];
    const int b = blockIdx.y;
    const int c = blockIdx.z;
    const int d = blockIdx.x * blockDim.x + threadIdx.x;

    float Av[D_STATE_], h[D_STATE_], p[D_STATE_];
#pragma unroll
    for (int n = 0; n < D_STATE_; ++n) {
        Av[n] = -__expf(A_log[(size_t)d * D_STATE_ + n]);
        h[n]  = 0.0f;
        p[n]  = 1.0f;
    }

    const int t0 = c * CH_LEN_;
    for (int tt = 0; tt < CH_LEN_; ++tt) {
        const size_t row = (size_t)(b * L_SZ + t0 + tt);
        if (threadIdx.x < 4) {  // 4 lanes x b128 = 64B = 16 floats of B_t
            uint32_t ldsa = (uint32_t)(uintptr_t)&sB[threadIdx.x * 4];
            const float* g = xdbl + row * XP_DIM_ + DT_RANK_ + threadIdx.x * 4;
            asm volatile("global_load_async_to_lds_b128 %0, %1, off"
                         :: "v"(ldsa), "v"(g) : "memory");
        }
        asm volatile("s_wait_asynccnt 0" ::: "memory");
        __syncthreads();

        const float dl = delta[row * D_INNER_ + d];
        const float du = dl * (float)u16[row * D_INNER_ + d];
#pragma unroll
        for (int n = 0; n < D_STATE_; ++n) {
            const float dA = __expf(dl * Av[n]);
            h[n] = dA * h[n] + du * sB[n];
            p[n] *= dA;
        }
        __syncthreads();
    }

#pragma unroll
    for (int n = 0; n < D_STATE_; ++n) {
        const size_t o = ((size_t)(c * B_SZ + b) * D_STATE_ + n) * D_INNER_ + d;
        Pbuf[o] = p[n];
        Hbuf[o] = h[n];
    }
}

// ---------------------------------------------------------------- selective scan, pass 2
// Sequentially combine the 16 chunk operators (h = P_c*h + h_c), then apply
// C_{L-1}, the u*D skip, and the silu(z) gate. Only t = L-1 output is needed
// (out[:, -1, :] is the reference's only consumer).
__global__ __launch_bounds__(256) void scan_pass2(const float* __restrict__ Pbuf,
                                                  const float* __restrict__ Hbuf,
                                                  const float* __restrict__ xdbl,
                                                  const bf16_t* __restrict__ u16,
                                                  const float* __restrict__ zlast,
                                                  const float* __restrict__ Dp,
                                                  float* __restrict__ ylast) {
    __shared__ float sC[D_STATE_];
    const int b = blockIdx.y;
    const int d = blockIdx.x * blockDim.x + threadIdx.x;
    if (threadIdx.x < D_STATE_)
        sC[threadIdx.x] = xdbl[(size_t)(b * L_SZ + (L_SZ - 1)) * XP_DIM_ +
                               DT_RANK_ + D_STATE_ + threadIdx.x];
    __syncthreads();

    float h[D_STATE_];
#pragma unroll
    for (int n = 0; n < D_STATE_; ++n) h[n] = 0.0f;

    for (int c = 0; c < NCHUNK_; ++c) {
#pragma unroll
        for (int n = 0; n < D_STATE_; ++n) {
            const size_t o = ((size_t)(c * B_SZ + b) * D_STATE_ + n) * D_INNER_ + d;
            h[n] = Pbuf[o] * h[n] + Hbuf[o];
        }
    }

    float y = 0.0f;
#pragma unroll
    for (int n = 0; n < D_STATE_; ++n) y += h[n] * sC[n];

    const float ul = (float)u16[(size_t)(b * L_SZ + (L_SZ - 1)) * D_INNER_ + d];
    const float z  = zlast[(size_t)b * D_INNER_ + d];
    ylast[(size_t)b * D_INNER_ + d] = (y + ul * Dp[d]) * silu_f(z);
}

// ---------------------------------------------------------------- out = y_last @ W_out^T  (8x1024)
__global__ __launch_bounds__(256) void outlast_kernel(const float* __restrict__ ylast,
                                                      const float* __restrict__ W_out,
                                                      float* __restrict__ olast) {
    int idx = blockIdx.x * blockDim.x + threadIdx.x;     // B*D_MODEL
    if (idx >= B_SZ * D_MODEL_) return;
    const int m = idx & (D_MODEL_ - 1);
    const int b = idx >> 10;
    const float* y = ylast + (size_t)b * D_INNER_;
    const float* w = W_out + (size_t)m * D_INNER_;
    float acc = 0.0f;
#pragma unroll 4
    for (int d = 0; d < D_INNER_; ++d) acc += y[d] * w[d];
    olast[idx] = acc;
}

// ---------------------------------------------------------------- logits = out_last @ W_fc^T + b_fc
__global__ __launch_bounds__(256) void fc_kernel(const float* __restrict__ olast,
                                                 const float* __restrict__ W_fc,
                                                 const float* __restrict__ b_fc,
                                                 float* __restrict__ out) {
    int idx = blockIdx.x * blockDim.x + threadIdx.x;     // B*N_CLASSES
    if (idx >= B_SZ * N_CLS_) return;
    const int c = idx % N_CLS_;
    const int b = idx / N_CLS_;
    const float* o = olast + (size_t)b * D_MODEL_;
    const float* w = W_fc + (size_t)c * D_MODEL_;
    float acc = b_fc[c];
#pragma unroll 4
    for (int m = 0; m < D_MODEL_; ++m) acc += o[m] * w[m];
    out[idx] = acc;
}

// ---------------------------------------------------------------- host launcher
extern "C" void kernel_launch(void* const* d_in, const int* in_sizes, int n_in,
                              void* d_out, int out_size, void* d_ws, size_t ws_size,
                              hipStream_t stream) {
    const float* x      = (const float*)d_in[0];
    const float* W_emb  = (const float*)d_in[1];
    const float* b_emb  = (const float*)d_in[2];
    const float* W_in   = (const float*)d_in[3];
    const float* W_conv = (const float*)d_in[4];
    const float* b_conv = (const float*)d_in[5];
    const float* W_xprj = (const float*)d_in[6];
    const float* W_dt   = (const float*)d_in[7];
    const float* b_dt   = (const float*)d_in[8];
    const float* A_log  = (const float*)d_in[9];
    const float* Dp     = (const float*)d_in[10];
    const float* W_out  = (const float*)d_in[11];
    const float* W_fc   = (const float*)d_in[12];
    const float* b_fc   = (const float*)d_in[13];
    float* out = (float*)d_out;

    char*  ws  = (char*)d_ws;
    size_t off = 0;
    auto alloc = [&](size_t bytes) -> void* {
        off = (off + 255) & ~(size_t)255;
        void* p = ws + off;
        off += bytes;
        return p;
    };

    bf16_t* x16    = (bf16_t*)alloc((size_t)M_ROWS * IN_DIM_ * 2);
    bf16_t* Wemb16 = (bf16_t*)alloc((size_t)D_MODEL_ * IN_DIM_ * 2);
    bf16_t* WinA16 = (bf16_t*)alloc((size_t)D_INNER_ * D_MODEL_ * 2);  // xs half only
    bf16_t* Wxp16  = (bf16_t*)alloc((size_t)XP_DIM_ * D_INNER_ * 2);
    bf16_t* Wdt16  = (bf16_t*)alloc((size_t)D_INNER_ * DT_RANK_ * 2);
    bf16_t* emb16  = (bf16_t*)alloc((size_t)M_ROWS * D_MODEL_ * 2);
    float*  xs     = (float*) alloc((size_t)M_ROWS * D_INNER_ * 4);
    bf16_t* u16    = (bf16_t*)alloc((size_t)M_ROWS * D_INNER_ * 2);
    float*  xdbl   = (float*) alloc((size_t)M_ROWS * XP_DIM_ * 4);
    bf16_t* dt16   = (bf16_t*)alloc((size_t)M_ROWS * DT_RANK_ * 2);
    float*  delta  = (float*) alloc((size_t)M_ROWS * D_INNER_ * 4);
    float*  Pbuf   = (float*) alloc((size_t)NCHUNK_ * B_SZ * D_STATE_ * D_INNER_ * 4);
    float*  Hbuf   = (float*) alloc((size_t)NCHUNK_ * B_SZ * D_STATE_ * D_INNER_ * 4);
    float*  zlast  = (float*) alloc((size_t)B_SZ * D_INNER_ * 4);
    float*  ylast  = (float*) alloc((size_t)B_SZ * D_INNER_ * 4);
    float*  olast  = (float*) alloc((size_t)B_SZ * D_MODEL_ * 4);

    auto cdiv = [](size_t a, size_t b) { return (unsigned)((a + b - 1) / b); };

    // ---- weight/activation conversion to bf16
    cvt_f32_bf16<<<cdiv((size_t)M_ROWS * IN_DIM_, 256), 256, 0, stream>>>(
        x, x16, (size_t)M_ROWS * IN_DIM_);
    cvt_f32_bf16<<<cdiv((size_t)D_MODEL_ * IN_DIM_, 256), 256, 0, stream>>>(
        W_emb, Wemb16, (size_t)D_MODEL_ * IN_DIM_);
    cvt_f32_bf16<<<cdiv((size_t)D_INNER_ * D_MODEL_, 256), 256, 0, stream>>>(
        W_in, WinA16, (size_t)D_INNER_ * D_MODEL_);   // rows [0, 2048): the xs half
    cvt_f32_bf16<<<cdiv((size_t)XP_DIM_ * D_INNER_, 256), 256, 0, stream>>>(
        W_xprj, Wxp16, (size_t)XP_DIM_ * D_INNER_);
    cvt_f32_bf16<<<cdiv((size_t)D_INNER_ * DT_RANK_, 256), 256, 0, stream>>>(
        W_dt, Wdt16, (size_t)D_INNER_ * DT_RANK_);

    // ---- GEMM1: emb16 = bf16(x @ W_emb^T + b_emb)   [8192 x 1024 x 256], 32x64 tiles
    wmma_gemm_nt<bf16_t, true, 0, 4>
        <<<cdiv((size_t)(M_ROWS / 32) * (D_MODEL_ / 64), 8), 256, 0, stream>>>(
            x16, Wemb16, b_emb, emb16, M_ROWS, D_MODEL_, IN_DIM_);

    // ---- GEMM2 (xs half only): xs = emb @ W_in[:2048]^T   [8192 x 2048 x 1024]
    wmma_gemm_nt<float, false, 0, 4>
        <<<cdiv((size_t)(M_ROWS / 32) * (D_INNER_ / 64), 8), 256, 0, stream>>>(
            emb16, WinA16, nullptr, xs, M_ROWS, D_INNER_, D_MODEL_);

    // ---- z needed only at t = L-1
    zlast_kernel<<<cdiv((size_t)B_SZ * D_INNER_, 256), 256, 0, stream>>>(
        emb16, W_in, zlast);

    // ---- depthwise causal conv + SiLU -> u (bf16)
    conv_silu_kernel<<<cdiv((size_t)M_ROWS * D_INNER_, 256), 256, 0, stream>>>(
        xs, W_conv, b_conv, u16);

    // ---- GEMM3: x_dbl = u @ W_xproj^T   [8192 x 96 x 2048], 32x32 tiles (96 = 3*32)
    wmma_gemm_nt<float, false, 0, 2>
        <<<cdiv((size_t)(M_ROWS / 32) * (XP_DIM_ / 32), 8), 256, 0, stream>>>(
            u16, Wxp16, nullptr, xdbl, M_ROWS, XP_DIM_, D_INNER_);

    // ---- dt slice -> bf16
    extract_dt_kernel<<<cdiv((size_t)M_ROWS * DT_RANK_, 256), 256, 0, stream>>>(
        xdbl, dt16);

    // ---- GEMM4: delta = softplus(dt @ W_dt^T + b_dt)   [8192 x 2048 x 64]
    wmma_gemm_nt<float, true, 1, 4>
        <<<cdiv((size_t)(M_ROWS / 32) * (D_INNER_ / 64), 8), 256, 0, stream>>>(
            dt16, Wdt16, b_dt, delta, M_ROWS, D_INNER_, DT_RANK_);

    // ---- chunked selective scan: 16 parallel chunks of 64 steps, then combine
    dim3 g1(D_INNER_ / 256, B_SZ, NCHUNK_);
    scan_pass1<<<g1, 256, 0, stream>>>(delta, xdbl, u16, A_log, Pbuf, Hbuf);
    dim3 g2(D_INNER_ / 256, B_SZ);
    scan_pass2<<<g2, 256, 0, stream>>>(Pbuf, Hbuf, xdbl, u16, zlast, Dp, ylast);

    // ---- out projection on last timestep only + classifier head
    outlast_kernel<<<cdiv((size_t)B_SZ * D_MODEL_, 256), 256, 0, stream>>>(
        ylast, W_out, olast);
    fc_kernel<<<cdiv((size_t)B_SZ * N_CLS_, 256), 256, 0, stream>>>(
        olast, W_fc, b_fc, out);
}
